// AttentionLSTM_3401614098619
// MI455X (gfx1250) — compile-verified, matrix operand order checked by reading, not run on hardware
//
#include <hip/hip_runtime.h>

// ============================================================================
// AttentionLSTM for MI455X (gfx1250): all GEMMs via v_wmma_f32_16x16x32_bf16
// (2x2 super-tiles per wave, hoisted fast/edge paths, b128 vector loads),
// persistent-workgroup WMMA LSTM with bf16 weights, h staged in LDS and c in
// VGPRs.
// ============================================================================

typedef __attribute__((ext_vector_type(16))) __bf16 v16bf;
typedef __attribute__((ext_vector_type(8)))  __bf16 v8bf;
typedef __attribute__((ext_vector_type(8)))  float  v8f;

#define DEV __device__ __forceinline__

constexpr int B_  = 32;
constexpr int L_  = 2160;
constexpr int D_  = 200;
constexpr int H_  = 768;
constexpr int T_  = 240;
constexpr int HD_ = 192;
constexpr int PF_ = 3072;
constexpr int NC_ = 2;
constexpr int H2_ = 1536;   // 2H
constexpr int G4_ = 3072;   // 4H
constexpr float NEGV = -1e10f;

// ---------------------------------------------------------------- fragments
DEV v8f zero8(){ v8f z;
#pragma unroll
  for (int i=0;i<8;i++) z[i]=0.f;
  return z; }

DEV v8f wmma_bf16(v16bf a, v16bf b, v8f c){
  return __builtin_amdgcn_wmma_f32_16x16x32_bf16(false, a, false, b, (short)0, c, false, false);
}

// Fast row-major loader: no bounds logic, 4x global_load_b128 + packed cvt.
DEV v16bf ld_rm_fast(const float* __restrict__ A, int ld, int r0, int k0){
  int l = threadIdx.x & 31; int r = l & 15; int hi = l >> 4;
  const float* p = A + (size_t)(r0 + r)*ld + k0 + hi*8;
  float4 x0 = *(const float4*)(p);
  float4 x1 = *(const float4*)(p + 4);
  float4 x2 = *(const float4*)(p + 16);
  float4 x3 = *(const float4*)(p + 20);
  v16bf f;
  f[0]=(__bf16)x0.x;  f[1]=(__bf16)x0.y;  f[2]=(__bf16)x0.z;  f[3]=(__bf16)x0.w;
  f[4]=(__bf16)x1.x;  f[5]=(__bf16)x1.y;  f[6]=(__bf16)x1.z;  f[7]=(__bf16)x1.w;
  f[8]=(__bf16)x2.x;  f[9]=(__bf16)x2.y;  f[10]=(__bf16)x2.z; f[11]=(__bf16)x2.w;
  f[12]=(__bf16)x3.x; f[13]=(__bf16)x3.y; f[14]=(__bf16)x3.z; f[15]=(__bf16)x3.w;
  return f;
}

// Fast column-major loader (W stored K x N): unguarded strided b32 loads.
DEV v16bf ld_cm_fast(const float* __restrict__ A, int ld, int n0, int k0){
  int l = threadIdx.x & 31; int r = l & 15; int hi = l >> 4;
  const float* p = A + (n0 + r) + (size_t)(k0 + hi*8)*ld;
  v16bf f;
#pragma unroll
  for (int j=0;j<8;j++)  f[j]   = (__bf16)p[(size_t)j*ld];
#pragma unroll
  for (int j=0;j<8;j++)  f[8+j] = (__bf16)p[(size_t)(16+j)*ld];
  return f;
}

// Guarded loaders (edge tiles and K-tails only).
DEV v16bf ld_rm_safe(const float* __restrict__ A, int ld, int rows, int cols,
                     int r0, int k0){
  int l = threadIdx.x & 31; int r = l & 15; int hi = l >> 4;
  int row = r0 + r; bool rok = row < rows;
  const float* p = A + (size_t)row*ld;
  v16bf f;
#pragma unroll
  for (int j=0;j<16;j++){
    int k = k0 + hi*8 + ((j<8)? j : (j+8));
    float v = (rok && k < cols) ? p[k] : 0.f;
    f[j] = (__bf16)v;
  }
  return f;
}

DEV v16bf ld_cm_safe(const float* __restrict__ A, int ld, int nsz, int ksz,
                     int n0, int k0){
  int l = threadIdx.x & 31; int r = l & 15; int hi = l >> 4;
  int n = n0 + r; bool nok = n < nsz;
  v16bf f;
#pragma unroll
  for (int j=0;j<16;j++){
    int k = k0 + hi*8 + ((j<8)? j : (j+8));
    float v = (nok && k < ksz) ? A[(size_t)k*ld + n] : 0.f;
    f[j] = (__bf16)v;
  }
  return f;
}

// bf16 fragment (row-major N x 768, always in bounds) — 2x b128 loads.
DEV v16bf ld_frag_wbf(const __bf16* __restrict__ A, int r0, int k0){
  int l = threadIdx.x & 31; int r = l & 15; int hi = l >> 4;
  const __bf16* p = A + (size_t)(r0 + r)*H_ + k0 + hi*8;
  v8bf x0 = *(const v8bf*)(p);
  v8bf x1 = *(const v8bf*)(p + 16);
  v16bf f;
#pragma unroll
  for (int j=0;j<8;j++){ f[j] = x0[j]; f[8+j] = x1[j]; }
  return f;
}

// bf16 fragment from LDS (row-major 32 x 768) — 2x ds_load_b128.
DEV v16bf ld_frag_lds(const __bf16* A, int r0, int k0){
  int l = threadIdx.x & 31; int r = l & 15; int hi = l >> 4;
  const __bf16* p = A + (r0 + r)*H_ + k0 + hi*8;
  v8bf x0 = *(const v8bf*)(p);
  v8bf x1 = *(const v8bf*)(p + 16);
  v16bf f;
#pragma unroll
  for (int j=0;j<8;j++){ f[j] = x0[j]; f[8+j] = x1[j]; }
  return f;
}

DEV float sigmoidf_(float x){ return 1.f/(1.f+expf(-x)); }

DEV float epilogue_val(float v, float scale, bool masked, float bias, int act){
  v *= scale;
  if (masked) v = NEGV;
  v += bias;
  if (act == 1)      v = tanhf(v);
  else if (act == 2) v = 0.5f*v*(1.f + erff(v*0.70710678118654752f));
  return v;
}

// ---------------------------------------------------------------- generic GEMM
// C[z,y] = act( scale*(A @ W^T|W) [mask->NEG] + b1 + b2 ).
// One 32x32 super-tile (2x2 WMMA tiles) per wave; 8 waves per block.
// Fast/edge paths selected once per wave; hot loop is load_b128 -> cvt -> wmma.
__global__ __launch_bounds__(256) void k_gemm(
    const float* __restrict__ A, int lda, long long sAy, long long sAz,
    const float* __restrict__ W, int ldw, long long sWy, long long sWz, int BT,
    float* __restrict__ C, int ldc, long long sCy, long long sCz,
    int M, int N, int K,
    const float* __restrict__ b1, const float* __restrict__ b2,
    float scale, const float* __restrict__ mask, int act)
{
  int wv = threadIdx.x >> 5;
  long long tile = (long long)blockIdx.x*8 + wv;
  int ntN = (N+31) >> 5;
  long long total = (long long)((M+31) >> 5) * ntN;
  if (tile >= total) return;
  int m0 = (int)(tile / ntN) * 32, n0 = (int)(tile % ntN) * 32;
  const float* Ab = A + (long long)blockIdx.y*sAy + (long long)blockIdx.z*sAz;
  const float* Wb = W + (long long)blockIdx.y*sWy + (long long)blockIdx.z*sWz;
  float*       Cb = C + (long long)blockIdx.y*sCy + (long long)blockIdx.z*sCz;

  v8f acc[2][2];
  acc[0][0]=zero8(); acc[0][1]=zero8(); acc[1][0]=zero8(); acc[1][1]=zero8();

  bool fullT = (m0 + 32 <= M) && (n0 + 32 <= N);
  int Kfull = K & ~31;

  if (fullT){
    if (BT){
      for (int k0=0;k0<Kfull;k0+=32){
        v16bf a0 = ld_rm_fast(Ab, lda, m0,    k0);
        v16bf a1 = ld_rm_fast(Ab, lda, m0+16, k0);
        v16bf b0 = ld_rm_fast(Wb, ldw, n0,    k0);
        v16bf b1f= ld_rm_fast(Wb, ldw, n0+16, k0);
        acc[0][0] = wmma_bf16(a0, b0,  acc[0][0]);
        acc[0][1] = wmma_bf16(a0, b1f, acc[0][1]);
        acc[1][0] = wmma_bf16(a1, b0,  acc[1][0]);
        acc[1][1] = wmma_bf16(a1, b1f, acc[1][1]);
      }
    } else {
      for (int k0=0;k0<Kfull;k0+=32){
        v16bf a0 = ld_rm_fast(Ab, lda, m0,    k0);
        v16bf a1 = ld_rm_fast(Ab, lda, m0+16, k0);
        v16bf b0 = ld_cm_fast(Wb, ldw, n0,    k0);
        v16bf b1f= ld_cm_fast(Wb, ldw, n0+16, k0);
        acc[0][0] = wmma_bf16(a0, b0,  acc[0][0]);
        acc[0][1] = wmma_bf16(a0, b1f, acc[0][1]);
        acc[1][0] = wmma_bf16(a1, b0,  acc[1][0]);
        acc[1][1] = wmma_bf16(a1, b1f, acc[1][1]);
      }
    }
    if (Kfull < K){                                   // K tail (rows full)
      int k0 = Kfull;
      v16bf a0 = ld_rm_safe(Ab, lda, M, K, m0,    k0);
      v16bf a1 = ld_rm_safe(Ab, lda, M, K, m0+16, k0);
      v16bf b0, b1f;
      if (BT){ b0 = ld_rm_safe(Wb, ldw, N, K, n0,    k0);
               b1f= ld_rm_safe(Wb, ldw, N, K, n0+16, k0); }
      else   { b0 = ld_cm_safe(Wb, ldw, N, K, n0,    k0);
               b1f= ld_cm_safe(Wb, ldw, N, K, n0+16, k0); }
      acc[0][0] = wmma_bf16(a0, b0,  acc[0][0]);
      acc[0][1] = wmma_bf16(a0, b1f, acc[0][1]);
      acc[1][0] = wmma_bf16(a1, b0,  acc[1][0]);
      acc[1][1] = wmma_bf16(a1, b1f, acc[1][1]);
    }
  } else {
    for (int k0=0;k0<K;k0+=32){
      v16bf a0 = ld_rm_safe(Ab, lda, M, K, m0,    k0);
      v16bf a1 = ld_rm_safe(Ab, lda, M, K, m0+16, k0);
      v16bf b0, b1f;
      if (BT){ b0 = ld_rm_safe(Wb, ldw, N, K, n0,    k0);
               b1f= ld_rm_safe(Wb, ldw, N, K, n0+16, k0); }
      else   { b0 = ld_cm_safe(Wb, ldw, N, K, n0,    k0);
               b1f= ld_cm_safe(Wb, ldw, N, K, n0+16, k0); }
      acc[0][0] = wmma_bf16(a0, b0,  acc[0][0]);
      acc[0][1] = wmma_bf16(a0, b1f, acc[0][1]);
      acc[1][0] = wmma_bf16(a1, b0,  acc[1][0]);
      acc[1][1] = wmma_bf16(a1, b1f, acc[1][1]);
    }
  }

  int l = threadIdx.x & 31; int colL = l & 15; int hi = l >> 4;
#pragma unroll
  for (int sj=0; sj<2; sj++){
    int n = n0 + sj*16 + colL;
    float bias = 0.f; bool masked = false;
    if (n < N){
      if (b1) bias += b1[n];
      if (b2) bias += b2[n];
      if (mask) masked = (mask[(size_t)blockIdx.z*N + n] == 0.f);
    }
    if (fullT){
#pragma unroll
      for (int si=0; si<2; si++){
        float* cp = Cb + (size_t)(m0 + si*16 + 8*hi)*ldc + n;
#pragma unroll
        for (int r=0;r<8;r++)
          cp[(size_t)r*ldc] = epilogue_val(acc[si][sj][r], scale, masked, bias, act);
      }
    } else {
#pragma unroll
      for (int si=0; si<2; si++){
#pragma unroll
        for (int r=0;r<8;r++){
          int m = m0 + si*16 + r + 8*hi;
          if (m < M && n < N)
            Cb[(size_t)m*ldc + n] = epilogue_val(acc[si][sj][r], scale, masked, bias, act);
        }
      }
    }
  }
}

// ---------------------------------------------------------------- weight cvt
__global__ void k_cvt(const float* __restrict__ in, __bf16* __restrict__ out, int n)
{
  int i = blockIdx.x*blockDim.x + threadIdx.x;
  if (i < n) out[i] = (__bf16)in[i];
}

// ---------------------------------------------------------------- LSTM
// One persistent block per direction. h: LDS bf16 (WMMA A source). c: VGPRs.
// Per wave / per jt: both M sub-tiles x 4 gate tiles share fragment loads.
__global__ __launch_bounds__(768) void k_lstm(
    const float* __restrict__ xWf, const float* __restrict__ xWb,
    const __bf16* __restrict__ Whhf, const __bf16* __restrict__ Whhb,
    float* __restrict__ out, float* __restrict__ hT, int Tlen)
{
  __shared__ __bf16 hbuf[32*H_];                 // 48 KB
  const int dir = blockIdx.x;
  const float*  xW  = dir ? xWb  : xWf;
  const __bf16* Whh = dir ? Whhb : Whhf;
  int tid = threadIdx.x;
  int wave = tid >> 5;
  int l = tid & 31; int colL = l & 15; int hi = l >> 4;

  for (int i=tid; i<32*H_; i+=768) hbuf[i] = (__bf16)0.f;
  v8f c_st[2][2]; v8f h_st[2][2];
#pragma unroll
  for (int a=0;a<2;a++)
#pragma unroll
    for (int b=0;b<2;b++) c_st[a][b] = zero8();
  __syncthreads();

  for (int t=0;t<Tlen;t++){
    const float* xwrow = xW + (size_t)t*32*G4_;
#pragma unroll
    for (int jj=0;jj<2;jj++){
      int jt = wave*2 + jj;
      v8f acc[2][4];
#pragma unroll
      for (int m=0;m<2;m++)
#pragma unroll
        for (int g=0;g<4;g++) acc[m][g] = zero8();
      for (int kc=0;kc<24;kc++){
        int k0 = kc*32;
        v16bf a0 = ld_frag_lds(hbuf, 0,  k0);
        v16bf a1 = ld_frag_lds(hbuf, 16, k0);
        v16bf w0 = ld_frag_wbf(Whh, 0*H_ + jt*16, k0);
        v16bf w1 = ld_frag_wbf(Whh, 1*H_ + jt*16, k0);
        v16bf w2 = ld_frag_wbf(Whh, 2*H_ + jt*16, k0);
        v16bf w3 = ld_frag_wbf(Whh, 3*H_ + jt*16, k0);
        acc[0][0] = wmma_bf16(a0, w0, acc[0][0]);
        acc[1][0] = wmma_bf16(a1, w0, acc[1][0]);
        acc[0][1] = wmma_bf16(a0, w1, acc[0][1]);
        acc[1][1] = wmma_bf16(a1, w1, acc[1][1]);
        acc[0][2] = wmma_bf16(a0, w2, acc[0][2]);
        acc[1][2] = wmma_bf16(a1, w2, acc[1][2]);
        acc[0][3] = wmma_bf16(a0, w3, acc[0][3]);
        acc[1][3] = wmma_bf16(a1, w3, acc[1][3]);
      }
      int jh = jt*16 + colL;
      int t_out = dir ? (Tlen-1-t) : t;
#pragma unroll
      for (int m=0;m<2;m++){
        v8f hnew;
#pragma unroll
        for (int r=0;r<8;r++){
          int b_ = m*16 + r + 8*hi;
          const float* xr = xwrow + (size_t)b_*G4_;
          float gi = acc[m][0][r] + xr[jh];
          float gf = acc[m][1][r] + xr[H_   + jh];
          float gg = acc[m][2][r] + xr[2*H_ + jh];
          float go = acc[m][3][r] + xr[3*H_ + jh];
          float cn = sigmoidf_(gf)*c_st[jj][m][r] + sigmoidf_(gi)*tanhf(gg);
          c_st[jj][m][r] = cn;
          float hv = sigmoidf_(go)*tanhf(cn);
          hnew[r] = hv;
          out[((size_t)b_*Tlen + t_out)*H2_ + dir*H_ + jh] = hv;
          if (hT && t == Tlen-1) hT[((size_t)dir*32 + b_)*H_ + jh] = hv;
        }
        h_st[jj][m] = hnew;
      }
    }
    __syncthreads();               // all reads of hbuf for step t done
#pragma unroll
    for (int jj=0;jj<2;jj++){
      int jh = (wave*2+jj)*16 + colL;
#pragma unroll
      for (int m=0;m<2;m++)
#pragma unroll
        for (int r=0;r<8;r++){
          int b_ = m*16 + r + 8*hi;
          hbuf[b_*H_ + jh] = (__bf16)h_st[jj][m][r];
        }
    }
    __syncthreads();               // new h visible
  }
}

// ---------------------------------------------------------------- elementwise
__global__ void k_embed(const int* __restrict__ x, const float* __restrict__ emb,
                        const float* __restrict__ pos,
                        float* __restrict__ xsf, float* __restrict__ xsb)
{
  int idx = blockIdx.x*blockDim.x + threadIdx.x;
  if (idx >= B_*T_*D_) return;
  int d = idx % D_; int t = (idx / D_) % T_; int b = idx / (D_*T_);
  const float sq = 14.142135623730951f;  // sqrt(200)
  float best = -3.4e38f;
#pragma unroll
  for (int j=0;j<3;j++){
    float acc = 0.f;
#pragma unroll
    for (int k=0;k<3;k++){
      int p = t*9 + j*3 + k;
      int tok = x[(size_t)b*L_ + p];
      acc += emb[(size_t)tok*D_ + d]*sq + pos[(size_t)p*D_ + d];
    }
    acc *= (1.f/3.f);
    best = fmaxf(best, acc);
  }
  xsf[((size_t)t*B_ + b)*D_ + d]          = best;
  xsb[((size_t)(T_-1-t)*B_ + b)*D_ + d]   = best;
}

__global__ void k_mask(const int* __restrict__ x, float* __restrict__ mask)
{
  int idx = blockIdx.x*blockDim.x + threadIdx.x;
  if (idx >= B_*T_) return;
  int t = idx % T_, b = idx / T_;
  float best = -3.4e38f;
#pragma unroll
  for (int j=0;j<3;j++){
    float acc = 0.f;
#pragma unroll
    for (int k=0;k<3;k++) acc += (float)x[(size_t)b*L_ + t*9 + j*3 + k];
    best = fmaxf(best, acc*(1.f/3.f));
  }
  mask[idx] = (best != 0.f) ? 1.f : 0.f;
}

__global__ void k_trans(const float* __restrict__ in,
                        float* __restrict__ xsf, float* __restrict__ xsb)
{ // (B,T,D) -> (T,B,D) and reversed copy
  int idx = blockIdx.x*blockDim.x + threadIdx.x;
  if (idx >= B_*T_*D_) return;
  int d = idx % D_; int t = (idx / D_) % T_; int b = idx / (D_*T_);
  float v = in[((size_t)b*T_ + t)*D_ + d];
  xsf[((size_t)t*B_ + b)*D_ + d]        = v;
  xsb[((size_t)(T_-1-t)*B_ + b)*D_ + d] = v;
}

__global__ __launch_bounds__(256) void k_ln(const float* __restrict__ x,
                                            const float* __restrict__ res,
                                            const float* __restrict__ g,
                                            const float* __restrict__ b,
                                            float* __restrict__ out, int W)
{
  __shared__ float sh[256];
  size_t row = blockIdx.x; int tid = threadIdx.x;
  const float* xr = x + row*W;
  const float* rr = res ? res + row*W : nullptr;
  float s = 0.f;
  for (int c=tid;c<W;c+=256) s += xr[c] + (rr?rr[c]:0.f);
  sh[tid]=s; __syncthreads();
  for (int k=128;k>0;k>>=1){ if(tid<k) sh[tid]+=sh[tid+k]; __syncthreads(); }
  float mean = sh[0]/W; __syncthreads();
  float s2 = 0.f;
  for (int c=tid;c<W;c+=256){ float v = xr[c] + (rr?rr[c]:0.f) - mean; s2 += v*v; }
  sh[tid]=s2; __syncthreads();
  for (int k=128;k>0;k>>=1){ if(tid<k) sh[tid]+=sh[tid+k]; __syncthreads(); }
  float inv = rsqrtf(sh[0]/W + 1e-5f); __syncthreads();
  for (int c=tid;c<W;c+=256){
    float v = xr[c] + (rr?rr[c]:0.f);
    out[row*W + c] = (v-mean)*inv*g[c] + b[c];
  }
}

__global__ __launch_bounds__(256) void k_softmax(const float* __restrict__ in,
                                                 float* __restrict__ out, int N)
{
  __shared__ float sh[256];
  size_t row = blockIdx.x; int tid = threadIdx.x;
  const float* xr = in + row*N;
  float mx = -3.4e38f;
  for (int c=tid;c<N;c+=256) mx = fmaxf(mx, xr[c]);
  sh[tid]=mx; __syncthreads();
  for (int k=128;k>0;k>>=1){ if(tid<k) sh[tid]=fmaxf(sh[tid],sh[tid+k]); __syncthreads(); }
  mx = sh[0]; __syncthreads();
  float s = 0.f;
  for (int c=tid;c<N;c+=256) s += expf(xr[c]-mx);
  sh[tid]=s; __syncthreads();
  for (int k=128;k>0;k>>=1){ if(tid<k) sh[tid]+=sh[tid+k]; __syncthreads(); }
  float inv = 1.f/sh[0]; __syncthreads();
  for (int c=tid;c<N;c+=256) out[row*N + c] = expf(xr[c]-mx)*inv;
}

__global__ __launch_bounds__(256) void k_kl(const float* __restrict__ yu,
                                            const float* __restrict__ e1,
                                            float* __restrict__ partials)
{
  __shared__ float sh[256];
  int bi = blockIdx.x;
  int b = bi / T_, i = bi % T_;
  int tid = threadIdx.x;
  const float* er = e1 + ((size_t)b*T_ + i)*T_;
  float mx = -3.4e38f;
  for (int j=tid;j<T_;j+=256) mx = fmaxf(mx, er[j]);
  sh[tid]=mx; __syncthreads();
  for (int k=128;k>0;k>>=1){ if(tid<k) sh[tid]=fmaxf(sh[tid],sh[tid+k]); __syncthreads(); }
  mx = sh[0]; __syncthreads();
  float s = 0.f;
  for (int j=tid;j<T_;j+=256) s += expf(er[j]-mx);
  sh[tid]=s; __syncthreads();
  for (int k=128;k>0;k>>=1){ if(tid<k) sh[tid]+=sh[tid+k]; __syncthreads(); }
  float lse = mx + logf(sh[0]); __syncthreads();
  float yi = yu[(size_t)b*T_ + i];
  float sd = 0.f;
  for (int j=tid;j<=i;j+=256) sd += expf(-fabsf(yi - yu[(size_t)b*T_ + j])*10.f);
  sh[tid]=sd; __syncthreads();
  for (int k=128;k>0;k>>=1){ if(tid<k) sh[tid]+=sh[tid+k]; __syncthreads(); }
  float lsd = logf(sh[0]); float invsd = 1.f/sh[0]; __syncthreads();
  float term = 0.f;
  for (int j=tid;j<=i;j+=256){
    float d  = -fabsf(yi - yu[(size_t)b*T_ + j])*10.f;
    float pe = expf(d)*invsd;
    if (pe > 0.f) term += pe*((d - lsd) - (er[j] - lse));
  }
  sh[tid]=term; __syncthreads();
  for (int k=128;k>0;k>>=1){ if(tid<k) sh[tid]+=sh[tid+k]; __syncthreads(); }
  if (tid==0) partials[bi] = sh[0];
}

__global__ __launch_bounds__(256) void k_klreduce(const float* __restrict__ partials,
                                                  float* __restrict__ kl)
{
  __shared__ float sh[256];
  float s = 0.f;
  for (int i=threadIdx.x;i<B_*T_;i+=256) s += partials[i];
  sh[threadIdx.x]=s; __syncthreads();
  for (int k=128;k>0;k>>=1){ if(threadIdx.x<k) sh[threadIdx.x]+=sh[threadIdx.x+k]; __syncthreads(); }
  if (threadIdx.x==0) kl[0] = sh[0] / (float)(T_*B_);
}

__global__ __launch_bounds__(256) void k_dot(const float* __restrict__ out2,
                                             const float* __restrict__ h1,
                                             float* __restrict__ sc)
{
  __shared__ float sh[256];
  int t = blockIdx.x, b = blockIdx.y, tid = threadIdx.x;
  const float* orow = out2 + ((size_t)b*T_ + t)*H2_;
  const float* hrow = h1 + (size_t)b*H2_;
  float s = 0.f;
  for (int d=tid; d<H2_; d+=256) s += orow[d]*hrow[d];
  sh[tid]=s; __syncthreads();
  for (int k=128;k>0;k>>=1){ if(tid<k) sh[tid]+=sh[tid+k]; __syncthreads(); }
  if (tid==0) sc[(size_t)b*T_ + t] = sh[0];
}

__global__ void k_pool(const float* __restrict__ out2, const float* __restrict__ aw,
                       float* __restrict__ pooled)
{
  int idx = blockIdx.x*blockDim.x + threadIdx.x;
  if (idx >= B_*H2_) return;
  int b = idx / H2_, d = idx % H2_;
  float s = 0.f;
  for (int t=0;t<T_;t++) s += out2[((size_t)b*T_ + t)*H2_ + d]*aw[(size_t)b*T_ + t];
  pooled[idx] = s;
}

__global__ void k_final(const float* __restrict__ m3, const float* __restrict__ kl,
                        float* __restrict__ out)
{
  int b = threadIdx.x;
  if (b < B_){
    float a = m3[b*2], c = m3[b*2+1];
    float mx = fmaxf(a,c);
    float ea = expf(a-mx), ec = expf(c-mx);
    float z = ea + ec;
    out[b*2]   = ea/z;
    out[b*2+1] = ec/z;
  }
  if (b == 0) out[B_*NC_] = kl[0];
}

// ---------------------------------------------------------------- host side
// workspace layout (in floats)
constexpr size_t F_MASK  = 0;                                  // B*T
constexpr size_t F_OUT1  = F_MASK + (size_t)B_*T_;             // B*T*2H
constexpr size_t F_SRC   = F_OUT1 + (size_t)B_*T_*H2_;         // B*T*H
constexpr size_t F_E1    = F_SRC  + (size_t)B_*T_*H_;          // B*T*T
constexpr size_t F_HT    = F_E1   + (size_t)B_*T_*T_;          // 2*B*H
constexpr size_t F_OUT2  = F_HT   + (size_t)2*B_*H_;           // B*T*2H
constexpr size_t F_SC    = F_OUT2 + (size_t)B_*T_*H2_;         // B*T
constexpr size_t F_AW    = F_SC   + (size_t)B_*T_;             // B*T
constexpr size_t F_POOL  = F_AW   + (size_t)B_*T_;             // B*2H
constexpr size_t F_M1    = F_POOL + (size_t)B_*H2_;            // B*4H
constexpr size_t F_M2    = F_M1   + (size_t)B_*G4_;            // B*2H
constexpr size_t F_M3    = F_M2   + (size_t)B_*H2_;            // B*NC (pad 64)
constexpr size_t F_KL    = F_M3   + 64;                        // 8
constexpr size_t F_KLP   = F_KL   + 8;                         // B*T partials
constexpr size_t F_OUT8  = F_KLP  + (size_t)B_*T_;             // B*T*D
constexpr size_t F_ARENA = F_OUT8 + (size_t)B_*T_*D_;
// arena sub-offsets — LSTM phase
constexpr size_t A_XSF = 0;
constexpr size_t A_XSB = A_XSF + (size_t)T_*B_*D_;
constexpr size_t A_XWF = A_XSB + (size_t)T_*B_*D_;
constexpr size_t A_XWB = A_XWF + (size_t)T_*B_*G4_;
constexpr size_t ARENA_SZ = A_XWB + (size_t)T_*B_*G4_;         // 50,257,920 floats
// arena sub-offsets — attention phase (reuses the same arena)
constexpr size_t A_Q   = 0;
constexpr size_t A_K   = A_Q   + (size_t)B_*T_*H_;
constexpr size_t A_V   = A_K   + (size_t)B_*T_*H_;
constexpr size_t A_EM  = A_V   + (size_t)B_*T_*H_;             // B*3*T*T
constexpr size_t A_X   = A_EM  + (size_t)B_*3*T_*T_;
constexpr size_t A_TMP = A_X   + (size_t)B_*T_*H_;
constexpr size_t A_Q1W = A_TMP + (size_t)B_*T_*H_;             // B*T*HD
constexpr size_t A_P1  = A_Q1W + (size_t)B_*T_*HD_;            // B*T*T
constexpr size_t A_FFH = 0;                                    // aliases Q..X (dead)
constexpr size_t A_FF2 = A_TMP;                                // aliases TMP (dead)
// bf16 LSTM weights (as float-slot offsets; 2 * 4H*H bf16 = 2,359,296 float slots)
constexpr size_t F_WBF   = F_ARENA + ARENA_SZ;

static void gemm(hipStream_t st,
                 const float* A, int lda, long long sAy, long long sAz,
                 const float* W, int ldw, long long sWy, long long sWz, int BT,
                 float* C, int ldc, long long sCy, long long sCz,
                 int M, int N, int K,
                 const float* b1, const float* b2,
                 float scale, const float* mask, int act,
                 int gy, int gz)
{
  long long tiles = (long long)((M+31)/32) * ((N+31)/32);
  dim3 grid((unsigned)((tiles + 7)/8), (unsigned)gy, (unsigned)gz);
  k_gemm<<<grid, 256, 0, st>>>(A, lda, sAy, sAz, W, ldw, sWy, sWz, BT,
                               C, ldc, sCy, sCz, M, N, K, b1, b2, scale, mask, act);
}

extern "C" void kernel_launch(void* const* d_in, const int* in_sizes, int n_in,
                              void* d_out, int out_size, void* d_ws, size_t ws_size,
                              hipStream_t stream)
{
  (void)in_sizes; (void)n_in; (void)out_size; (void)ws_size;
  auto FP = [&](int i){ return (const float*)d_in[i]; };
  const int* x   = (const int*)d_in[0];
  const float* yu = FP(1);
  // 2 emb, 3 pos, 4..11 lstm, 12 attn_W, 13 attn_b, 14 fc7_W, 15 fc7_b,
  // 16..32 layer0, 33..49 layer1, 50..61 tail.
  float* w  = (float*)d_ws;
  float* ar = w + F_ARENA;
  __bf16* whf = (__bf16*)(w + F_WBF);
  __bf16* whb = whf + (size_t)G4_*H_;
  const float iscl = 1.0f / sqrtf((float)HD_);

  // ---- LSTM recurrent weights -> bf16 (once per launch) ----
  k_cvt<<<(G4_*H_ + 255)/256, 256, 0, stream>>>(FP(5), whf, G4_*H_);
  k_cvt<<<(G4_*H_ + 255)/256, 256, 0, stream>>>(FP(9), whb, G4_*H_);

  // ---- phase 1: embedding + pooling + mask ----
  k_embed<<<(B_*T_*D_ + 255)/256, 256, 0, stream>>>(x, FP(2), FP(3), ar+A_XSF, ar+A_XSB);
  k_mask <<<(B_*T_ + 255)/256,    256, 0, stream>>>(x, w + F_MASK);

  // ---- BiLSTM #1: xW precompute + persistent recurrence ----
  gemm(stream, ar+A_XSF, D_,0,0, FP(4), D_,0,0, 1, ar+A_XWF, G4_,0,0,
       T_*B_, G4_, D_, FP(6), FP(7), 1.f, nullptr, 0, 1,1);
  gemm(stream, ar+A_XSB, D_,0,0, FP(8), D_,0,0, 1, ar+A_XWB, G4_,0,0,
       T_*B_, G4_, D_, FP(10), FP(11), 1.f, nullptr, 0, 1,1);
  k_lstm<<<2, 768, 0, stream>>>(ar+A_XWF, ar+A_XWB, whf, whb,
                                w + F_OUT1, nullptr, T_);

  // ---- attn_W tanh + fc7 -> src ----
  gemm(stream, w+F_OUT1, H2_,0,0, FP(12), H2_,0,0, 1, ar+A_Q, H2_,0,0,
       B_*T_, H2_, H2_, FP(13), nullptr, 1.f, nullptr, 1, 1,1);     // tanh, in arena
  gemm(stream, ar+A_Q, H2_,0,0, FP(14), H2_,0,0, 1, w+F_SRC, H_,0,0,
       B_*T_, H_, H2_, FP(15), nullptr, 1.f, nullptr, 0, 1,1);

  // ---- 2 encoder layers ----
  for (int li=0; li<2; li++){
    int P = 16 + li*17;
    // QKV projections (B*T, H)
    gemm(stream, w+F_SRC, H_,0,0, FP(P+0), H_,0,0, 1, ar+A_Q, H_,0,0,
         B_*T_, H_, H_, FP(P+1), nullptr, 1.f, nullptr, 0, 1,1);
    gemm(stream, w+F_SRC, H_,0,0, FP(P+2), H_,0,0, 1, ar+A_K, H_,0,0,
         B_*T_, H_, H_, FP(P+3), nullptr, 1.f, nullptr, 0, 1,1);
    gemm(stream, w+F_SRC, H_,0,0, FP(P+4), H_,0,0, 1, ar+A_V, H_,0,0,
         B_*T_, H_, H_, FP(P+5), nullptr, 1.f, nullptr, 0, 1,1);
    // scores for heads 0..2: em[b,h] = mask(Q K^T / sqrt(HD))
    gemm(stream, ar+A_Q, H_, HD_, (long long)T_*H_,
                 ar+A_K, H_, HD_, (long long)T_*H_, 1,
                 ar+A_EM, T_, (long long)T_*T_, (long long)3*T_*T_,
         T_, T_, HD_, nullptr, nullptr, iscl, w+F_MASK, 0, 3, B_);
    // special head: Q1w = Q1 @ rw[b]   (rw NOT transposed)
    gemm(stream, ar+A_Q + 3*HD_, H_, 0, (long long)T_*H_,
                 FP(P+16), HD_, 0, (long long)HD_*HD_, 0,
                 ar+A_Q1W, HD_, 0, (long long)T_*HD_,
         T_, HD_, HD_, nullptr, nullptr, 1.f, nullptr, 0, 1, B_);
    // e1 = mask(Q1w K1^T / sqrt(HD))   (kept for KL)
    gemm(stream, ar+A_Q1W, HD_, 0, (long long)T_*HD_,
                 ar+A_K + 3*HD_, H_, 0, (long long)T_*H_, 1,
                 w+F_E1, T_, 0, (long long)T_*T_,
         T_, T_, HD_, nullptr, nullptr, iscl, w+F_MASK, 0, 1, B_);
    // softmaxes
    k_softmax<<<B_*3*T_, 256, 0, stream>>>(ar+A_EM, ar+A_EM, T_);
    k_softmax<<<B_*T_,   256, 0, stream>>>(w+F_E1,  ar+A_P1, T_);
    // xm = P @ V (heads 0..2), written into X at (b,t,h*HD+d)
    gemm(stream, ar+A_EM, T_, (long long)T_*T_, (long long)3*T_*T_,
                 ar+A_V, H_, HD_, (long long)T_*H_, 0,
                 ar+A_X, H_, HD_, (long long)T_*H_,
         T_, HD_, T_, nullptr, nullptr, 1.f, nullptr, 0, 3, B_);
    // x1 = P1 @ V1 into X head 3
    gemm(stream, ar+A_P1, T_, 0, (long long)T_*T_,
                 ar+A_V + 3*HD_, H_, 0, (long long)T_*H_, 0,
                 ar+A_X + 3*HD_, H_, 0, (long long)T_*H_,
         T_, HD_, T_, nullptr, nullptr, 1.f, nullptr, 0, 1, B_);
    // out-proj + residual LN
    gemm(stream, ar+A_X, H_,0,0, FP(P+6), H_,0,0, 1, ar+A_TMP, H_,0,0,
         B_*T_, H_, H_, FP(P+7), nullptr, 1.f, nullptr, 0, 1,1);
    k_ln<<<B_*T_, 256, 0, stream>>>(w+F_SRC, ar+A_TMP, FP(P+8), FP(P+9), w+F_SRC, H_);
    // FF: GELU(ff1) -> ff2 -> residual LN
    gemm(stream, w+F_SRC, H_,0,0, FP(P+10), H_,0,0, 1, ar+A_FFH, PF_,0,0,
         B_*T_, PF_, H_, FP(P+11), nullptr, 1.f, nullptr, 2, 1,1);
    gemm(stream, ar+A_FFH, PF_,0,0, FP(P+12), PF_,0,0, 1, ar+A_FF2, H_,0,0,
         B_*T_, H_, PF_, FP(P+13), nullptr, 1.f, nullptr, 0, 1,1);
    k_ln<<<B_*T_, 256, 0, stream>>>(w+F_SRC, ar+A_FF2, FP(P+14), FP(P+15), w+F_SRC, H_);
  }
  // ---- feature KL from layer-1 e1 ----
  k_kl<<<B_*T_, 256, 0, stream>>>(yu, w+F_E1, w+F_KLP);
  k_klreduce<<<1, 256, 0, stream>>>(w+F_KLP, w+F_KL);

  // ---- fc6 tanh -> LN -> fc8 ----
  gemm(stream, w+F_SRC, H_,0,0, FP(50), H_,0,0, 1, ar, H_,0,0,
       B_*T_, H_, H_, FP(51), nullptr, 1.f, nullptr, 1, 1,1);
  k_ln<<<B_*T_, 256, 0, stream>>>(ar, nullptr, FP(52), FP(53), w+F_SRC, H_);
  gemm(stream, w+F_SRC, H_,0,0, FP(54), H_,0,0, 1, w+F_OUT8, D_,0,0,
       B_*T_, D_, H_, FP(55), nullptr, 1.f, nullptr, 0, 1,1);

  // ---- BiLSTM #2 ----
  k_trans<<<(B_*T_*D_ + 255)/256, 256, 0, stream>>>(w+F_OUT8, ar+A_XSF, ar+A_XSB);
  gemm(stream, ar+A_XSF, D_,0,0, FP(4), D_,0,0, 1, ar+A_XWF, G4_,0,0,
       T_*B_, G4_, D_, FP(6), FP(7), 1.f, nullptr, 0, 1,1);
  gemm(stream, ar+A_XSB, D_,0,0, FP(8), D_,0,0, 1, ar+A_XWB, G4_,0,0,
       T_*B_, G4_, D_, FP(10), FP(11), 1.f, nullptr, 0, 1,1);
  k_lstm<<<2, 768, 0, stream>>>(ar+A_XWF, ar+A_XWB, whf, whb,
                                w + F_OUT2, w + F_HT, T_);

  // ---- attention pooling: h1 is just hT flat (reshape is layout-preserving) ----
  k_dot<<<dim3(T_, B_), 256, 0, stream>>>(w+F_OUT2, w+F_HT, w+F_SC);
  k_softmax<<<B_, 256, 0, stream>>>(w+F_SC, w+F_AW, T_);
  k_pool<<<(B_*H2_ + 255)/256, 256, 0, stream>>>(w+F_OUT2, w+F_AW, w+F_POOL);

  // ---- MLP head + final softmax + kl ----
  gemm(stream, w+F_POOL, H2_,0,0, FP(56), H2_,0,0, 1, w+F_M1, G4_,0,0,
       B_, G4_, H2_, FP(57), nullptr, 1.f, nullptr, 0, 1,1);
  gemm(stream, w+F_M1, G4_,0,0, FP(58), G4_,0,0, 1, w+F_M2, H2_,0,0,
       B_, H2_, G4_, FP(59), nullptr, 1.f, nullptr, 0, 1,1);
  gemm(stream, w+F_M2, H2_,0,0, FP(60), H2_,0,0, 1, w+F_M3, NC_,0,0,
       B_, NC_, H2_, FP(61), nullptr, 1.f, nullptr, 0, 1,1);
  k_final<<<1, 64, 0, stream>>>(w+F_M3, w+F_KL, (float*)d_out);
}
